// TFFactorAtt_33217277067351
// MI455X (gfx1250) — matrix operand order repaired and checked by reading, not compiled, not run on hardware
//
#include <hip/hip_runtime.h>
#include <hip/hip_bf16.h>
#include <math.h>

// ---------------------------------------------------------------------------
// Shapes
// ---------------------------------------------------------------------------
#define Bz   32
#define Nn   785
#define DIM  768
#define Hh   8
#define HD   96
#define HW   28

typedef __attribute__((ext_vector_type(16))) __bf16 v16bf;
typedef __attribute__((ext_vector_type(8)))  float  v8f;

union FragBF { v16bf v; unsigned int u[8]; };

#define TILE_M 128
#define TILE_N 64
#define TILE_K 32
#define LDSK   34   // 34 bf16 = 68B = 17 dwords -> conflict-free row stride

__device__ __forceinline__ void cvt_split(float x, __bf16& hi, __bf16& lo) {
    hi = (__bf16)x;
    lo = (__bf16)(x - (float)hi);
}

// ---------------------------------------------------------------------------
// Kernel 1/5: C = A(MxK) @ B(KxN) [+ bias], fp32 in/out, bf16x3 split WMMA
// grid = (N/64, ceil(M/128)), block = 256 (8 waves, 4x2 of 32x32 tiles)
// ---------------------------------------------------------------------------
__global__ __launch_bounds__(256)
void gemm_bf16x3(const float* __restrict__ A, const float* __restrict__ Bm,
                 const float* __restrict__ bias, float* __restrict__ C,
                 int M, int N, int K)
{
    __shared__ __bf16 Ah[TILE_M * LDSK];
    __shared__ __bf16 Al[TILE_M * LDSK];
    __shared__ __bf16 Bh[TILE_N * LDSK];
    __shared__ __bf16 Bl[TILE_N * LDSK];

    const int tid   = threadIdx.x;
    const int wid   = tid >> 5;
    const int lane  = tid & 31;
    const int khalf = lane >> 4;
    const int laneR = lane & 15;
    const int waveM = wid & 3;   // 0..3  -> 32-row strip
    const int waveN = wid >> 2;  // 0..1  -> 32-col strip

    const int mBase = blockIdx.y * TILE_M;
    const int nBase = blockIdx.x * TILE_N;

    v8f acc[2][2];
    v8f zero = {};
    acc[0][0] = zero; acc[0][1] = zero; acc[1][0] = zero; acc[1][1] = zero;

    for (int k0 = 0; k0 < K; k0 += TILE_K) {
        // ---- stage A tile (128x32) with fp32->bf16 hi/lo split ----
        #pragma unroll
        for (int i = 0; i < 4; ++i) {
            int r  = (tid >> 3) + 32 * i;
            int c4 = (tid & 7) * 4;
            int g  = mBase + r;
            float4 val = make_float4(0.f, 0.f, 0.f, 0.f);
            if (g < M) val = *(const float4*)(A + (size_t)g * K + k0 + c4);
            float vv[4] = {val.x, val.y, val.z, val.w};
            #pragma unroll
            for (int j = 0; j < 4; ++j) {
                __bf16 h, l; cvt_split(vv[j], h, l);
                Ah[r * LDSK + c4 + j] = h;
                Al[r * LDSK + c4 + j] = l;
            }
        }
        // ---- stage B tile (32x64), transposed into LDS as [n][k] ----
        #pragma unroll
        for (int i = 0; i < 2; ++i) {
            int kk = (tid >> 4) + 16 * i;
            int c4 = (tid & 15) * 4;
            float4 val = *(const float4*)(Bm + (size_t)(k0 + kk) * N + nBase + c4);
            float vv[4] = {val.x, val.y, val.z, val.w};
            #pragma unroll
            for (int j = 0; j < 4; ++j) {
                __bf16 h, l; cvt_split(vv[j], h, l);
                Bh[(c4 + j) * LDSK + kk] = h;
                Bl[(c4 + j) * LDSK + kk] = l;
            }
        }
        __syncthreads();

        // ---- build fragments per ISA 7.12.2 layouts ----
        FragBF ahf[2], alf[2], bhf[2], blf[2];
        #pragma unroll
        for (int s = 0; s < 2; ++s) {
            int row = waveM * 32 + s * 16 + laneR;
            #pragma unroll
            for (int p = 0; p < 8; ++p) {
                int ka = ((p >= 4) ? 16 : 0) + khalf * 8 + (p & 3) * 2;
                ahf[s].u[p] = *(const unsigned int*)&Ah[row * LDSK + ka];
                alf[s].u[p] = *(const unsigned int*)&Al[row * LDSK + ka];
            }
        }
        #pragma unroll
        for (int t = 0; t < 2; ++t) {
            int col = waveN * 32 + t * 16 + laneR;
            #pragma unroll
            for (int p = 0; p < 8; ++p) {
                int kb = ((p >= 4) ? 16 : 0) + khalf * 8 + (p & 3) * 2;
                bhf[t].u[p] = *(const unsigned int*)&Bh[col * LDSK + kb];
                blf[t].u[p] = *(const unsigned int*)&Bl[col * LDSK + kb];
            }
        }

        // ---- 3-product bf16 split accumulate (Ah*Bh + Ah*Bl + Al*Bh) ----
        #pragma unroll
        for (int s = 0; s < 2; ++s) {
            #pragma unroll
            for (int t = 0; t < 2; ++t) {
                acc[s][t] = __builtin_amdgcn_wmma_f32_16x16x32_bf16(
                    false, ahf[s].v, false, bhf[t].v, (short)0, acc[s][t], false, false);
                acc[s][t] = __builtin_amdgcn_wmma_f32_16x16x32_bf16(
                    false, ahf[s].v, false, blf[t].v, (short)0, acc[s][t], false, false);
                acc[s][t] = __builtin_amdgcn_wmma_f32_16x16x32_bf16(
                    false, alf[s].v, false, bhf[t].v, (short)0, acc[s][t], false, false);
            }
        }
        __syncthreads();
    }

    // ---- store D (VGPR r -> row r + khalf*8, col = laneR) ----
    #pragma unroll
    for (int s = 0; s < 2; ++s) {
        #pragma unroll
        for (int t = 0; t < 2; ++t) {
            int col = nBase + waveN * 32 + t * 16 + laneR;
            float bv = bias ? bias[col] : 0.f;
            #pragma unroll
            for (int r = 0; r < 8; ++r) {
                int row = mBase + waveM * 32 + s * 16 + khalf * 8 + r;
                if (row < M) C[(size_t)row * N + col] = acc[s][t][r] + bv;
            }
        }
    }
}

// ---------------------------------------------------------------------------
// Kernel 2: fused softmax(k, axis=N) and kTv[b,h] = ks^T @ v   (96x96)
// grid = (H, B), block = 96 threads; thread t owns v-column t
// ---------------------------------------------------------------------------
__global__ __launch_bounds__(96)
void softmax_ktv(const float* __restrict__ qkv, float* __restrict__ kTv)
{
    const int h = blockIdx.x;
    const int b = blockIdx.y;
    const int t = threadIdx.x;
    const size_t strideN = 3 * Hh * HD; // 2304
    const float* kb = qkv + (((size_t)b * Nn * 3 + 1) * Hh + h) * HD + t;
    const float* vb = qkv + (((size_t)b * Nn * 3 + 2) * Hh + h) * HD + t;

    float m = -3.0e38f;
    for (int n = 0; n < Nn; ++n) m = fmaxf(m, kb[(size_t)n * strideN]);
    float s = 0.f;
    for (int n = 0; n < Nn; ++n) s += __expf(kb[(size_t)n * strideN] - m);
    const float inv = 1.f / s;

    float acc[HD];
    #pragma unroll
    for (int i = 0; i < HD; ++i) acc[i] = 0.f;

    __shared__ float ksh[HD];
    for (int n = 0; n < Nn; ++n) {
        float kv = __expf(kb[(size_t)n * strideN] - m) * inv;
        float vv = vb[(size_t)n * strideN];
        __syncthreads();
        ksh[t] = kv;
        __syncthreads();
        #pragma unroll
        for (int kk = 0; kk < HD; ++kk) acc[kk] += ksh[kk] * vv;
    }
    float* out = kTv + ((size_t)(b * Hh + h) * HD) * HD + t;
    #pragma unroll
    for (int kk = 0; kk < HD; ++kk) out[(size_t)kk * HD] = acc[kk];
}

// ---------------------------------------------------------------------------
// Kernel 3: depthwise SAME conv on v_img, sliding window down y
// thread = (b, x, c-in-group); channels contiguous -> coalesced
// ---------------------------------------------------------------------------
template<int KS>
__global__ __launch_bounds__(256)
void dwconv(const float* __restrict__ qkv, const float* __restrict__ w,
            const float* __restrict__ bias, float* __restrict__ out,
            int cBase, int cCount)
{
    const int R = KS / 2;
    int idx = blockIdx.x * 256 + threadIdx.x;
    int total = Bz * HW * cCount;
    if (idx >= total) return;
    int cl = idx % cCount;
    int x  = (idx / cCount) % HW;
    int b  = idx / (cCount * HW);
    int c  = cBase + cl;

    float wreg[KS][KS];
    #pragma unroll
    for (int i = 0; i < KS; ++i)
        #pragma unroll
        for (int j = 0; j < KS; ++j)
            wreg[i][j] = w[(i * KS + j) * cCount + cl];
    const float bv = bias[cl];

    auto vload = [&](int yy, int xx) -> float {
        if (yy < 0 || yy >= HW || xx < 0 || xx >= HW) return 0.f;
        size_t n = 1 + (size_t)yy * HW + xx;
        return qkv[(((size_t)b * Nn + n) * 3 + 2) * DIM + c];
    };

    float win[KS][KS];
    #pragma unroll
    for (int i = 0; i < KS - 1; ++i) {
        int yy = i - R;
        #pragma unroll
        for (int j = 0; j < KS; ++j) win[i][j] = vload(yy, x + j - R);
    }
    for (int y = 0; y < HW; ++y) {
        #pragma unroll
        for (int j = 0; j < KS; ++j) win[KS - 1][j] = vload(y + R, x + j - R);
        float a = bv;
        #pragma unroll
        for (int i = 0; i < KS; ++i)
            #pragma unroll
            for (int j = 0; j < KS; ++j)
                a += win[i][j] * wreg[i][j];
        out[((size_t)b * (HW * HW) + (size_t)y * HW + x) * DIM + c] = a;
        #pragma unroll
        for (int i = 0; i < KS - 1; ++i)
            #pragma unroll
            for (int j = 0; j < KS; ++j) win[i][j] = win[i + 1][j];
    }
}

// ---------------------------------------------------------------------------
// Kernel 4: att = scale * (q @ kTv) + (n>0 ? q * conv_v : 0)
// grid = (ceil(N/64), H, B), block = 256; kTv + q tile in LDS
// ---------------------------------------------------------------------------
__global__ __launch_bounds__(256)
void combine_att(const float* __restrict__ qkv, const float* __restrict__ kTv,
                 const float* __restrict__ conv, float* __restrict__ att)
{
    const int nt = blockIdx.x, h = blockIdx.y, b = blockIdx.z;
    const int tid = threadIdx.x;
    __shared__ float kTvs[HD * HD];    // 36 KB
    __shared__ float qs[64 * HD];      // 24 KB

    const float* src = kTv + (size_t)(b * Hh + h) * HD * HD;
    for (int i = tid; i < HD * HD; i += 256) kTvs[i] = src[i];
    const int n0 = nt * 64;
    for (int i = tid; i < 64 * HD; i += 256) {
        int nl = i / HD, hd = i % HD;
        int n = n0 + nl;
        qs[i] = (n < Nn)
            ? qkv[((((size_t)b * Nn + n) * 3 + 0) * Hh + h) * HD + hd] : 0.f;
    }
    __syncthreads();

    const float scale = 0.10206207261596575f; // 96^-0.5
    for (int i = tid; i < 64 * HD; i += 256) {
        int nl = i / HD, hd = i % HD;
        int n = n0 + nl;
        if (n >= Nn) continue;
        float a = 0.f;
        #pragma unroll 8
        for (int kk = 0; kk < HD; ++kk) a += qs[nl * HD + kk] * kTvs[kk * HD + hd];
        a *= scale;
        if (n >= 1)
            a += qs[nl * HD + hd] *
                 conv[((size_t)b * (HW * HW) + (n - 1)) * DIM + h * HD + hd];
        att[((size_t)b * Nn + n) * DIM + h * HD + hd] = a;
    }
}

// ---------------------------------------------------------------------------
extern "C" void kernel_launch(void* const* d_in, const int* in_sizes, int n_in,
                              void* d_out, int out_size, void* d_ws, size_t ws_size,
                              hipStream_t stream) {
    const float* x      = (const float*)d_in[0];
    const float* qkv_w  = (const float*)d_in[1];
    const float* proj_w = (const float*)d_in[2];
    const float* proj_b = (const float*)d_in[3];
    const float* k3     = (const float*)d_in[4];
    const float* b3     = (const float*)d_in[5];
    const float* k5     = (const float*)d_in[6];
    const float* b5     = (const float*)d_in[7];
    const float* k7     = (const float*)d_in[8];
    const float* b7     = (const float*)d_in[9];
    float* out = (float*)d_out;

    float* ws   = (float*)d_ws;
    float* qkv  = ws;                                   // 32*785*2304
    float* kTv  = qkv  + (size_t)Bz * Nn * 3 * DIM;     // 32*8*96*96
    float* conv = kTv  + (size_t)Bz * Hh * HD * HD;     // 32*784*768
    float* att  = conv + (size_t)Bz * HW * HW * DIM;    // 32*785*768

    const int M = Bz * Nn; // 25120

    // 1) qkv = x @ qkv_w  (WMMA bf16x3)
    dim3 g1(3 * DIM / TILE_N, (M + TILE_M - 1) / TILE_M);
    gemm_bf16x3<<<g1, 256, 0, stream>>>(x, qkv_w, nullptr, qkv, M, 3 * DIM, DIM);

    // 2) softmax over N + kTv
    softmax_ktv<<<dim3(Hh, Bz), HD, 0, stream>>>(qkv, kTv);

    // 3) depthwise CRPE convs
    {
        int t3 = Bz * HW * 192;
        dwconv<3><<<(t3 + 255) / 256, 256, 0, stream>>>(qkv, k3, b3, conv, 0, 192);
        int t5 = Bz * HW * 288;
        dwconv<5><<<(t5 + 255) / 256, 256, 0, stream>>>(qkv, k5, b5, conv, 192, 288);
        int t7 = Bz * HW * 288;
        dwconv<7><<<(t7 + 255) / 256, 256, 0, stream>>>(qkv, k7, b7, conv, 480, 288);
    }

    // 4) att = scale*factor_att + crpe
    combine_att<<<dim3((Nn + 63) / 64, Hh, Bz), 256, 0, stream>>>(qkv, kTv, conv, att);

    // 5) out = att @ proj_w + proj_b  (WMMA bf16x3)
    dim3 g5(DIM / TILE_N, (M + TILE_M - 1) / TILE_M);
    gemm_bf16x3<<<g5, 256, 0, stream>>>(att, proj_w, proj_b, out, M, DIM, DIM);
}
